// GrobnerCritic_2199023255723
// MI455X (gfx1250) — matrix operand order repaired and checked by reading, not compiled, not run on hardware
//
#include <hip/hip_runtime.h>

typedef __attribute__((ext_vector_type(16))) __bf16 v16bf;
typedef __attribute__((ext_vector_type(8)))  __bf16 v8bf;
typedef __attribute__((ext_vector_type(8)))  float  v8f;

// ---------- helpers ----------
__device__ __forceinline__ __bf16 f2bf(float f) {
  unsigned u = __builtin_bit_cast(unsigned, f);
  unsigned r = u + 0x7fffu + ((u >> 16) & 1u);   // round-to-nearest-even
  unsigned short h = (unsigned short)(r >> 16);
  return __builtin_bit_cast(__bf16, h);
}
__device__ __forceinline__ float bf2f(__bf16 b) {
  unsigned short h = __builtin_bit_cast(unsigned short, b);
  unsigned u = ((unsigned)h) << 16;
  return __builtin_bit_cast(float, u);
}
__device__ __forceinline__ float vsum8(v8f t) {
  return ((t[0] + t[1]) + (t[2] + t[3])) + ((t[4] + t[5]) + (t[6] + t[7]));
}

#define WMMA_BF16(a, b, c) \
  __builtin_amdgcn_wmma_f32_16x16x32_bf16(false, (a), false, (b), (short)0, (c), false, false)

// ---------- CDNA5 async global->LDS copy (ASYNCcnt-tracked, no VGPR bounce) ----------
// Per-lane: LDS[lds_off] = MEM[gaddr], 16 bytes. One instruction per wave.
__device__ __forceinline__ void async_copy_b128(unsigned lds_off, const void* gaddr) {
  asm volatile("global_load_async_to_lds_b128 %0, %1, off"
               :: "v"(lds_off), "v"(gaddr) : "memory");
}
template <int N>
__device__ __forceinline__ void wait_asynccnt() {
  asm volatile("s_wait_asynccnt %0" :: "i"(N) : "memory");
}
// Low 32 bits of a generic LDS pointer are the wave-relative LDS byte address.
__device__ __forceinline__ unsigned lds_addr(const void* p) {
  return (unsigned)(size_t)p;
}

// One 16x16 output tile, accumulating over K (LDK) in steps of 32.
// Hin: LDS bf16 [.. x LDK] row-major. Wg: global bf16 [O x LDK] row-major
// (torch Linear weight (out,in)), so B = W^T needs no transpose:
//   B lane n holds W[ct*16+n, k..k+15] contiguous.
template <int LDK>
__device__ __forceinline__ v8f tile_gemm(const __bf16* Hin, const __bf16* __restrict__ Wg,
                                         int rt, int ct, int lane, v8f acc) {
  const int half = lane >> 4;     // 0: lanes 0-15, 1: lanes 16-31
  const int l15  = lane & 15;
#pragma unroll
  for (int k0 = 0; k0 < LDK; k0 += 32) {
    // A: lane<16 -> row l15, K[k0..k0+15]; lane>=16 -> K[k0+16..k0+31]
    const __bf16* ar = Hin + (rt * 16 + l15) * LDK + k0 + 16 * half;
    v8bf alo = *(const v8bf*)ar;
    v8bf ahi = *(const v8bf*)(ar + 8);
    v16bf a = __builtin_shufflevector(alo, ahi, 0, 1, 2, 3, 4, 5, 6, 7,
                                      8, 9, 10, 11, 12, 13, 14, 15);
    // B: lane half selects K sub-block of 16, contiguous in weight row
    v16bf b = *(const v16bf*)(Wg + (size_t)(ct * 16 + l15) * LDK + k0 + 16 * half);
    acc = WMMA_BF16(a, b, acc);
  }
  return acc;
}

// C/D layout: VGPR v, lane l -> M = v + 8*(l/16), N = l%16
__device__ __forceinline__ void store_tile_lds_relu(__bf16* H, const v8f acc,
                                                    const float* __restrict__ bias,
                                                    int rt, int ct, int lane) {
  const int half = lane >> 4, l15 = lane & 15;
  const float bv = bias[ct * 16 + l15];
#pragma unroll
  for (int v = 0; v < 8; ++v) {
    float y = acc[v] + bv;
    y = fmaxf(y, 0.f);
    H[(rt * 16 + v + 8 * half) * 256 + ct * 16 + l15] = f2bf(y);
  }
}

// ---------- fused DeepSets phi: 3-layer MLP (in=K1 -> 256 -> 256 -> 256) over
// ROWS = 32*PASSES set elements per group, summed over the set dimension.
// Poly stage:  K1=32,  PASSES=2 (64 monomials), CONV=true  (f32 input)
// Ideal stage: K1=256, PASSES=1 (32 polys),     CONV=false (bf16 input, async staged)
template <int K1, int PASSES, bool CONV>
__global__ __launch_bounds__(256) void phi_sum_kernel(
    const float* __restrict__ xf, const __bf16* __restrict__ xb,
    const __bf16* __restrict__ W0, const float* __restrict__ b0,
    const __bf16* __restrict__ W1, const float* __restrict__ b1,
    const __bf16* __restrict__ W2, const float* __restrict__ b2,
    __bf16* __restrict__ out) {
  __shared__ __bf16 Xs[32 * K1];
  __shared__ __bf16 H1[32 * 256];
  __shared__ __bf16 H2[32 * 256];

  const int tid = threadIdx.x;
  const int wv = tid >> 5, lane = tid & 31;
  const int ct0 = wv, ct1 = wv + 8;   // each wave owns 2 column tiles x 2 row tiles
  const int g = blockIdx.x;
  const int ROWS = 32 * PASSES;
  float s0 = 0.f, s1 = 0.f;           // column sums (set reduction), in registers

  for (int pass = 0; pass < PASSES; ++pass) {
    const size_t base = ((size_t)g * ROWS + (size_t)pass * 32) * K1;
    if constexpr (CONV) {
      for (int i = tid * 4; i < 32 * K1; i += 1024) {
        float4 vv = *(const float4*)(xf + base + i);
        Xs[i + 0] = f2bf(vv.x); Xs[i + 1] = f2bf(vv.y);
        Xs[i + 2] = f2bf(vv.z); Xs[i + 3] = f2bf(vv.w);
      }
    } else {
      const unsigned xsBase = lds_addr(&Xs[0]);
      for (int i = tid * 8; i < 32 * K1; i += 2048) {
        async_copy_b128(xsBase + (unsigned)i * 2u, xb + base + i);
      }
      wait_asynccnt<0>();
    }
    __syncthreads();

    // ---- layer 1: Xs[32xK1] -> H1[32x256], ReLU
    {
      v8f t00 = {0,0,0,0,0,0,0,0}, t01 = {0,0,0,0,0,0,0,0};
      v8f t10 = {0,0,0,0,0,0,0,0}, t11 = {0,0,0,0,0,0,0,0};
      t00 = tile_gemm<K1>(Xs, W0, 0, ct0, lane, t00);
      t01 = tile_gemm<K1>(Xs, W0, 0, ct1, lane, t01);
      t10 = tile_gemm<K1>(Xs, W0, 1, ct0, lane, t10);
      t11 = tile_gemm<K1>(Xs, W0, 1, ct1, lane, t11);
      store_tile_lds_relu(H1, t00, b0, 0, ct0, lane);
      store_tile_lds_relu(H1, t01, b0, 0, ct1, lane);
      store_tile_lds_relu(H1, t10, b0, 1, ct0, lane);
      store_tile_lds_relu(H1, t11, b0, 1, ct1, lane);
    }
    __syncthreads();

    // ---- layer 2: H1 -> H2, ReLU
    {
      v8f t00 = {0,0,0,0,0,0,0,0}, t01 = {0,0,0,0,0,0,0,0};
      v8f t10 = {0,0,0,0,0,0,0,0}, t11 = {0,0,0,0,0,0,0,0};
      t00 = tile_gemm<256>(H1, W1, 0, ct0, lane, t00);
      t01 = tile_gemm<256>(H1, W1, 0, ct1, lane, t01);
      t10 = tile_gemm<256>(H1, W1, 1, ct0, lane, t10);
      t11 = tile_gemm<256>(H1, W1, 1, ct1, lane, t11);
      store_tile_lds_relu(H2, t00, b1, 0, ct0, lane);
      store_tile_lds_relu(H2, t01, b1, 0, ct1, lane);
      store_tile_lds_relu(H2, t10, b1, 1, ct0, lane);
      store_tile_lds_relu(H2, t11, b1, 1, ct1, lane);
    }
    __syncthreads();

    // ---- layer 3: H2 -> column sums (no activation; bias folded in at the end)
    {
      v8f t = {0,0,0,0,0,0,0,0};
      t = tile_gemm<256>(H2, W2, 0, ct0, lane, t);  s0 += vsum8(t);
      t = (v8f){0,0,0,0,0,0,0,0};
      t = tile_gemm<256>(H2, W2, 1, ct0, lane, t);  s0 += vsum8(t);
      t = (v8f){0,0,0,0,0,0,0,0};
      t = tile_gemm<256>(H2, W2, 0, ct1, lane, t);  s1 += vsum8(t);
      t = (v8f){0,0,0,0,0,0,0,0};
      t = tile_gemm<256>(H2, W2, 1, ct1, lane, t);  s1 += vsum8(t);
    }
    __syncthreads();
  }

  // lanes l and l+16 hold the two row-halves of the same column N = l%16
  s0 += __shfl_down(s0, 16, 32);
  s1 += __shfl_down(s1, 16, 32);
  if (lane < 16) {
    const int c0 = ct0 * 16 + lane, c1 = ct1 * 16 + lane;
    out[(size_t)g * 256 + c0] = f2bf(s0 + (float)ROWS * b2[c0]);
    out[(size_t)g * 256 + c1] = f2bf(s1 + (float)ROWS * b2[c1]);
  }
}

// ---------- generic Y[N,O] = act(X[N,K] * W[O,K]^T + b), bf16 in/out ----------
// Block: 64 rows x 64 cols, 8 waves, each wave -> one 16-row tile x two 16-col tiles.
// X tile staged to LDS via async global->LDS copies, double buffered.
__global__ __launch_bounds__(256) void gemm_bias_act_kernel(
    const __bf16* __restrict__ X, const __bf16* __restrict__ W,
    const float* __restrict__ bias, __bf16* __restrict__ Y,
    int K, int O, int relu) {
  __shared__ __bf16 Xs[2][64 * 32];
  const int tid = threadIdx.x;
  const int wv = tid >> 5, lane = tid & 31;
  const int half = lane >> 4, l15 = lane & 15;
  const int rt = wv >> 1;
  const int ctA = (wv & 1) * 2;
  const size_t rowBase = (size_t)blockIdx.y * 64;
  const int colBase = blockIdx.x * 64;
  const int lr = tid >> 2;              // 0..63: row staged by this thread
  const int lc = (tid & 3) * 8;         // 0,8,16,24: col chunk (8 bf16 = 16B)
  v8f acc0 = {0,0,0,0,0,0,0,0}, acc1 = {0,0,0,0,0,0,0,0};

  const unsigned ldsBuf0 = lds_addr(&Xs[0][0]);
  const unsigned ldsBuf1 = lds_addr(&Xs[1][0]);
  const unsigned myOff = (unsigned)((lr * 32 + lc) * 2);
  const __bf16* gsrcRow = X + (rowBase + lr) * K + lc;

  // prime the pipeline: chunk 0 -> buffer 0
  async_copy_b128(ldsBuf0 + myOff, gsrcRow);
  int cur = 0;
  for (int k0 = 0; k0 < K; k0 += 32) {
    if (k0 + 32 < K) {
      // overlap: fetch next chunk into the other buffer while we compute
      async_copy_b128((cur ? ldsBuf0 : ldsBuf1) + myOff, gsrcRow + k0 + 32);
      wait_asynccnt<1>();   // current buffer's copy complete (in-order), next in flight
    } else {
      wait_asynccnt<0>();
    }
    __syncthreads();        // all waves' contributions to Xs[cur] visible

    const __bf16* XsC = &Xs[cur][0];
    const __bf16* ar = XsC + (rt * 16 + l15) * 32 + 16 * half;
    v8bf alo = *(const v8bf*)ar;
    v8bf ahi = *(const v8bf*)(ar + 8);
    v16bf a = __builtin_shufflevector(alo, ahi, 0, 1, 2, 3, 4, 5, 6, 7,
                                      8, 9, 10, 11, 12, 13, 14, 15);
    const __bf16* bp = &W[(size_t)(colBase + ctA * 16 + l15) * K + k0 + 16 * half];
    v16bf b0 = *(const v16bf*)bp;
    v16bf b1 = *(const v16bf*)(bp + (size_t)16 * K);
    if (k0 + 32 < K) __builtin_prefetch(bp + 32, 0, 1);   // next K-chunk of W
    acc0 = WMMA_BF16(a, b0, acc0);
    acc1 = WMMA_BF16(a, b1, acc1);
    __syncthreads();        // all reads of Xs[cur] done before it is overwritten
    cur ^= 1;
  }

  const int c0 = colBase + ctA * 16 + l15;
  const int c1 = c0 + 16;
  const float bv0 = bias[c0], bv1 = bias[c1];
#pragma unroll
  for (int v = 0; v < 8; ++v) {
    const size_t r = rowBase + rt * 16 + v + 8 * half;
    float y0 = acc0[v] + bv0, y1 = acc1[v] + bv1;
    if (relu) { y0 = fmaxf(y0, 0.f); y1 = fmaxf(y1, 0.f); }
    Y[r * O + c0] = f2bf(y0);
    Y[r * O + c1] = f2bf(y1);
  }
}

// ---------- final 512 -> 1 head: one wave per batch row ----------
__global__ __launch_bounds__(256) void eval_final_kernel(
    const __bf16* __restrict__ H, const __bf16* __restrict__ w1,
    const float* __restrict__ b1, float* __restrict__ out) {
  const int wv = threadIdx.x >> 5, lane = threadIdx.x & 31;
  const int row = blockIdx.x * 8 + wv;
  const __bf16* hr = H + (size_t)row * 512;
  float s = 0.f;
  for (int i = lane * 8; i < 512; i += 256) {
    v8bf hv = *(const v8bf*)(hr + i);
    v8bf wb = *(const v8bf*)(w1 + i);
#pragma unroll
    for (int j = 0; j < 8; ++j) s += bf2f(hv[j]) * bf2f(wb[j]);
  }
#pragma unroll
  for (int off = 16; off > 0; off >>= 1) s += __shfl_down(s, off, 32);
  if (lane == 0) out[row] = s + b1[0];
}

// ---------- f32 -> bf16 weight conversion ----------
__global__ void f2bf_kernel(const float* __restrict__ in, __bf16* __restrict__ out, int n) {
  int i = blockIdx.x * 256 + threadIdx.x;
  if (i < n) out[i] = f2bf(in[i]);
}

extern "C" void kernel_launch(void* const* d_in, const int* in_sizes, int n_in,
                              void* d_out, int out_size, void* d_ws, size_t ws_size,
                              hipStream_t stream) {
  (void)n_in; (void)out_size; (void)ws_size;
  // Input order: x; poly{phi_w0..2, phi_b0..2, rho_w0..2, rho_b0..2};
  //              ideal{same}; eval{w0, b0, w1, b1}
  const float* x = (const float*)d_in[0];
  const float* ppw[3] = {(const float*)d_in[1],  (const float*)d_in[2],  (const float*)d_in[3]};
  const float* ppb[3] = {(const float*)d_in[4],  (const float*)d_in[5],  (const float*)d_in[6]};
  const float* prw[3] = {(const float*)d_in[7],  (const float*)d_in[8],  (const float*)d_in[9]};
  const float* prb[3] = {(const float*)d_in[10], (const float*)d_in[11], (const float*)d_in[12]};
  const float* ipw[3] = {(const float*)d_in[13], (const float*)d_in[14], (const float*)d_in[15]};
  const float* ipb[3] = {(const float*)d_in[16], (const float*)d_in[17], (const float*)d_in[18]};
  const float* irw[3] = {(const float*)d_in[19], (const float*)d_in[20], (const float*)d_in[21]};
  const float* irb[3] = {(const float*)d_in[22], (const float*)d_in[23], (const float*)d_in[24]};
  const float* ew0 = (const float*)d_in[25];
  const float* eb0 = (const float*)d_in[26];
  const float* ew1 = (const float*)d_in[27];
  const float* eb1 = (const float*)d_in[28];

  char* ws = (char*)d_ws;
  size_t off = 0;
  auto alloc = [&](size_t bytes) -> void* {
    off = (off + 255) & ~(size_t)255;
    void* p = ws + off;
    off += bytes;
    return p;
  };
  auto cvt = [&](const float* src, int n) -> __bf16* {
    __bf16* dst = (__bf16*)alloc((size_t)n * 2);
    f2bf_kernel<<<(n + 255) / 256, 256, 0, stream>>>(src, dst, n);
    return dst;
  };

  __bf16 *bppw[3], *bprw[3], *bipw[3], *birw[3];
  for (int i = 0; i < 3; ++i) bppw[i] = cvt(ppw[i], in_sizes[1 + i]);
  for (int i = 0; i < 3; ++i) bprw[i] = cvt(prw[i], in_sizes[7 + i]);
  for (int i = 0; i < 3; ++i) bipw[i] = cvt(ipw[i], in_sizes[13 + i]);
  for (int i = 0; i < 3; ++i) birw[i] = cvt(irw[i], in_sizes[19 + i]);
  __bf16* bew0 = cvt(ew0, in_sizes[25]);
  __bf16* bew1 = cvt(ew1, in_sizes[27]);

  const int NB = 128, BP = 4096;  // batch, batch*polys
  __bf16* S1 = (__bf16*)alloc((size_t)BP * 256 * 2);  // poly phi summed over monomials
  __bf16* T1 = (__bf16*)alloc((size_t)BP * 256 * 2);
  __bf16* T2 = (__bf16*)alloc((size_t)BP * 256 * 2);
  __bf16* T3 = (__bf16*)alloc((size_t)BP * 256 * 2);  // poly encodings
  __bf16* S2 = (__bf16*)alloc((size_t)NB * 256 * 2);  // ideal phi summed over polys
  __bf16* U1 = (__bf16*)alloc((size_t)NB * 512 * 2);
  __bf16* U2 = (__bf16*)alloc((size_t)NB * 512 * 2);
  __bf16* U3 = (__bf16*)alloc((size_t)NB * 512 * 2);  // ideal embedding
  __bf16* HV = (__bf16*)alloc((size_t)NB * 512 * 2);  // evaluator hidden

  // 1) poly phi (32->256->256->256) fused with sum over 64 monomials
  phi_sum_kernel<32, 2, true><<<BP, 256, 0, stream>>>(
      x, nullptr, bppw[0], ppb[0], bppw[1], ppb[1], bppw[2], ppb[2], S1);

  // 2) poly rho (256->256->256->256)
  gemm_bias_act_kernel<<<dim3(4, 64), 256, 0, stream>>>(S1, bprw[0], prb[0], T1, 256, 256, 1);
  gemm_bias_act_kernel<<<dim3(4, 64), 256, 0, stream>>>(T1, bprw[1], prb[1], T2, 256, 256, 1);
  gemm_bias_act_kernel<<<dim3(4, 64), 256, 0, stream>>>(T2, bprw[2], prb[2], T3, 256, 256, 0);

  // 3) ideal phi (256->256->256->256) fused with sum over 32 polys
  phi_sum_kernel<256, 1, false><<<NB, 256, 0, stream>>>(
      nullptr, T3, bipw[0], ipb[0], bipw[1], ipb[1], bipw[2], ipb[2], S2);

  // 4) ideal rho (256->512->512->512)
  gemm_bias_act_kernel<<<dim3(8, 2), 256, 0, stream>>>(S2, birw[0], irb[0], U1, 256, 512, 1);
  gemm_bias_act_kernel<<<dim3(8, 2), 256, 0, stream>>>(U1, birw[1], irb[1], U2, 512, 512, 1);
  gemm_bias_act_kernel<<<dim3(8, 2), 256, 0, stream>>>(U2, birw[2], irb[2], U3, 512, 512, 0);

  // 5) evaluator: ReLU(Linear 512->512), then 512->1
  gemm_bias_act_kernel<<<dim3(8, 2), 256, 0, stream>>>(U3, bew0, eb0, HV, 512, 512, 1);
  eval_final_kernel<<<16, 256, 0, stream>>>(HV, bew1, eb1, (float*)d_out);
}
